// VerifiableComputation_37864431681871
// MI455X (gfx1250) — compile-verified, matrix operand order checked by reading, not run on hardware
//
#include <hip/hip_runtime.h>
#include <hip/hip_bf16.h>
#include <stdint.h>
#include <stddef.h>

// ---------------------------------------------------------------------------
// Shapes
#define H   1024
#define H2  2048
#define H3  3072
#define H4  4096
#define TT  2048      // B*S tokens
#define SS  1024      // sequence length per batch
#define MMEM 512      // memory slots
#define NHD 16        // heads
#define DH  64        // head dim
#define NE  8         // experts

typedef unsigned short u16;
typedef __attribute__((ext_vector_type(16))) __bf16 v16bf;
typedef __attribute__((ext_vector_type(8)))  float  v8f;
typedef __attribute__((ext_vector_type(4)))  unsigned int u32x4;
typedef __attribute__((ext_vector_type(4)))  int  i32x4;
typedef __attribute__((ext_vector_type(8)))  int  i32x8;

#if defined(__has_builtin)
# if __has_builtin(__builtin_amdgcn_tensor_load_to_lds) && __has_builtin(__builtin_amdgcn_s_wait_tensorcnt)
#  define USE_TDM 1
# endif
#endif
#ifndef USE_TDM
# define USE_TDM 0
#endif

// ---------------------------------------------------------------------------
// Device helpers
__device__ __forceinline__ u16 f2bf(float f) {
  union { float f; unsigned u; } cv; cv.f = f;
  unsigned u = cv.u;
  u += 0x7fffu + ((u >> 16) & 1u);        // round-to-nearest-even
  return (u16)(u >> 16);
}
__device__ __forceinline__ float gelu_f(float x)    { return 0.5f * x * (1.0f + erff(x * 0.70710678f)); }
__device__ __forceinline__ float sigmoid_f(float x) { return 1.0f / (1.0f + expf(-x)); }
__device__ __forceinline__ float act_f(float v, int act) {
  if (act == 1) return gelu_f(v);
  if (act == 2) return fmaxf(v, 0.0f);
  if (act == 3) return sigmoid_f(v);
  return v;
}

#if USE_TDM
// Build a 2D TDM descriptor (bf16 elements) and issue TENSOR_LOAD_TO_LDS.
// Tile = tileRows x tileK elements, row stride = rowStride elements.
// rowsAvail/rowLen bound the tensor so OOB rows/cols are zero-filled by HW.
__device__ __forceinline__ void tdm_load_2d(unsigned ldsOff, const void* gp,
                                            unsigned rowLen, unsigned rowsAvail,
                                            unsigned rowStride, unsigned tileK,
                                            unsigned tileRows) {
  unsigned long long ga  = (unsigned long long)(size_t)gp;
  unsigned long long st0 = (unsigned long long)rowStride;  // elements
  u32x4 g0;
  g0[0] = 1u;                                            // count=1 (valid D#)
  g0[1] = ldsOff;                                        // lds_addr
  g0[2] = (unsigned)(ga & 0xffffffffull);                // global_addr[31:0]
  g0[3] = (unsigned)((ga >> 32) & 0x1ffffffull) | (2u << 30);  // addr[56:32] | type=2
  i32x8 g1;
  g1[0] = (int)(1u << 16);                               // data_size=2B, wg_mask=0
  g1[1] = (int)((rowLen & 0xffffu) << 16);               // tensor_dim0[15:0]
  g1[2] = (int)(((rowLen >> 16) & 0xffffu) | ((rowsAvail & 0xffffu) << 16));
  g1[3] = (int)(((rowsAvail >> 16) & 0xffffu) | ((tileK & 0xffffu) << 16));
  g1[4] = (int)(tileRows & 0xffffu);                     // tile_dim1, tile_dim2=0
  g1[5] = (int)(st0 & 0xffffffffull);                    // dim0_stride[31:0]
  g1[6] = (int)((st0 >> 32) & 0xffffull);                // dim0_stride[47:32]
  g1[7] = 0;
  i32x4 z4 = {0, 0, 0, 0};
#if defined(__clang_major__) && __clang_major__ >= 23
  i32x8 z8 = {0, 0, 0, 0, 0, 0, 0, 0};
  __builtin_amdgcn_tensor_load_to_lds(g0, g1, z4, z4, z8, 0);
#else
  __builtin_amdgcn_tensor_load_to_lds(g0, g1, z4, z4, 0);
#endif
}
#endif

// ---------------------------------------------------------------------------
// WMMA GEMM: C[M,N] = act(A[M,K] * B[N,K]^T + bias) [* rowScale] [+= / =]
// A,B bf16 (K contiguous), C f32, optional bf16 mirror Cbf.
// Block = 256 threads = 8 waves (4 x 2). Workgroup tile 128(M) x 128(N),
// wave tile 32 x 64 = 2x4 WMMA accumulators, K-step 32.
// LDS double-buffered; TDM software pipeline: issue stage i+1, wait tensorcnt<=2
// so stage i+1's DMA overlaps stage i's WMMA work.
#define BM 128
#define BN 128
#define BK 32

__global__ void __launch_bounds__(256)
wmma_gemm_kernel(const u16* __restrict__ A, const u16* __restrict__ Bw,
                 const float* __restrict__ bias,
                 float* __restrict__ C, u16* __restrict__ Cbf,
                 int Mr, int Nc, int K,
                 long long sA, long long sB, long long sC,
                 int act, const float* __restrict__ rowScale, int rsStride,
                 int accum) {
  __shared__ __align__(64) u16 ldsA[2][BM * BK];
  __shared__ __align__(64) u16 ldsB[2][BN * BK];
  const int tid  = threadIdx.x;
  const int bz   = blockIdx.z;
  const u16* Ab  = A + (long long)bz * sA;
  const u16* Bb  = Bw + (long long)bz * sB;
  const int mBase = blockIdx.y * BM;
  const int nBase = blockIdx.x * BN;
  const int lane = tid & 31;
  const int w    = tid >> 5;
  const int wm   = w >> 1;   // 0..3  (32 rows each)
  const int wn   = w & 1;    // 0..1  (64 cols each)

  v8f zero8 = {0.f, 0.f, 0.f, 0.f, 0.f, 0.f, 0.f, 0.f};
  v8f acc[2][4];
  #pragma unroll
  for (int i = 0; i < 2; ++i)
    #pragma unroll
    for (int j = 0; j < 4; ++j) acc[i][j] = zero8;

  const int nk = K / BK;     // K is always a multiple of 32 here

#if USE_TDM
  if (tid == 0) {            // prologue: stage 0 -> buffer 0
    tdm_load_2d((unsigned)(unsigned long long)(size_t)&ldsA[0][0],
                Ab + (long long)mBase * K, (unsigned)K, (unsigned)(Mr - mBase),
                (unsigned)K, BK, BM);
    tdm_load_2d((unsigned)(unsigned long long)(size_t)&ldsB[0][0],
                Bb + (long long)nBase * K, (unsigned)K, (unsigned)(Nc - nBase),
                (unsigned)K, BK, BN);
  }
#endif

  for (int it = 0; it < nk; ++it) {
    const int buf = it & 1;
    const int k0  = it * BK;
#if USE_TDM
    if (tid == 0) {
      if (it + 1 < nk) {     // issue next stage into the other buffer
        const int k1 = k0 + BK;
        tdm_load_2d((unsigned)(unsigned long long)(size_t)&ldsA[buf ^ 1][0],
                    Ab + (long long)mBase * K + k1, (unsigned)(K - k1),
                    (unsigned)(Mr - mBase), (unsigned)K, BK, BM);
        tdm_load_2d((unsigned)(unsigned long long)(size_t)&ldsB[buf ^ 1][0],
                    Bb + (long long)nBase * K + k1, (unsigned)(K - k1),
                    (unsigned)(Nc - nBase), (unsigned)K, BK, BN);
        __builtin_amdgcn_s_wait_tensorcnt(2);   // current stage's 2 loads done
      } else {
        __builtin_amdgcn_s_wait_tensorcnt(0);
      }
    }
#else
    for (int idx = tid; idx < BM * BK; idx += 256) {
      int r = idx >> 5, c = idx & 31;
      int row = mBase + r;
      ldsA[buf][idx] = (row < Mr) ? Ab[(long long)row * K + k0 + c] : (u16)0;
    }
    for (int idx = tid; idx < BN * BK; idx += 256) {
      int r = idx >> 5, c = idx & 31;
      int row = nBase + r;
      ldsB[buf][idx] = (row < Nc) ? Bb[(long long)row * K + k0 + c] : (u16)0;
    }
#endif
    __syncthreads();         // stage `it` resident in ldsX[buf]

    v16bf afr[2], bfr[4];
    #pragma unroll
    for (int i = 0; i < 2; ++i)
      afr[i] = *reinterpret_cast<const v16bf*>(
          &ldsA[buf][(wm * 32 + i * 16 + (lane & 15)) * BK + (lane >> 4) * 16]);
    #pragma unroll
    for (int j = 0; j < 4; ++j)
      bfr[j] = *reinterpret_cast<const v16bf*>(
          &ldsB[buf][(wn * 64 + j * 16 + (lane & 15)) * BK + (lane >> 4) * 16]);

    #pragma unroll
    for (int i = 0; i < 2; ++i)
      #pragma unroll
      for (int j = 0; j < 4; ++j) {
#if defined(__HIP_DEVICE_COMPILE__)
        acc[i][j] = __builtin_amdgcn_wmma_f32_16x16x32_bf16(
            false, afr[i], false, bfr[j], (short)0, acc[i][j], false, false);
#else
        (void)afr; (void)bfr;
#endif
      }
    __syncthreads();         // all waves done reading ldsX[buf] -> reusable
  }

  // Epilogue: C layout per ISA — VGPR r, lane l -> row r+8*(l>>4), col l&15.
  const int rloc = lane >> 4, cloc = lane & 15;
  #pragma unroll
  for (int i = 0; i < 2; ++i)
    #pragma unroll
    for (int j = 0; j < 4; ++j) {
      int col = nBase + wn * 64 + j * 16 + cloc;
      if (col >= Nc) continue;
      float bv = bias ? bias[col] : 0.0f;
      #pragma unroll
      for (int r = 0; r < 8; ++r) {
        int row = mBase + wm * 32 + i * 16 + r + 8 * rloc;
        if (row >= Mr) continue;
        float v = act_f(acc[i][j][r] + bv, act);
        if (rowScale) v *= rowScale[(long long)row * rsStride];
        long long idx = (long long)bz * sC + (long long)row * Nc + col;
        if (C)   { if (accum) C[idx] += v; else C[idx] = v; }
        if (Cbf) Cbf[idx] = f2bf(v);
      }
    }
}

// ---------------------------------------------------------------------------
// Elementwise / reduction kernels
__global__ void cvt_bf16_kernel(const float* __restrict__ in, u16* __restrict__ out,
                                long long n) {
  long long i = (long long)blockIdx.x * blockDim.x + threadIdx.x;
  long long st = (long long)gridDim.x * blockDim.x;
  for (; i < n; i += st) out[i] = f2bf(in[i]);
}

__global__ void zero_kernel(float* __restrict__ p, long long n) {
  long long i = (long long)blockIdx.x * blockDim.x + threadIdx.x;
  long long st = (long long)gridDim.x * blockDim.x;
  for (; i < n; i += st) p[i] = 0.0f;
}

// LayerNorm over `cols`, optional residual add and trailing GELU.
__global__ void ln_kernel(const float* __restrict__ x, const float* __restrict__ res,
                          const float* __restrict__ g, const float* __restrict__ b,
                          float* __restrict__ out, u16* __restrict__ obf,
                          int cols, int act) {
  __shared__ float vbuf[2048];
  __shared__ float red[256];
  const int row = blockIdx.x, tid = threadIdx.x;
  const float* xr = x + (long long)row * cols;
  const float* rr = res ? res + (long long)row * cols : nullptr;
  float s = 0.0f;
  for (int c = tid; c < cols; c += 256) {
    float v = xr[c] + (rr ? rr[c] : 0.0f);
    vbuf[c] = v; s += v;
  }
  red[tid] = s; __syncthreads();
  for (int o = 128; o > 0; o >>= 1) { if (tid < o) red[tid] += red[tid + o]; __syncthreads(); }
  float mean = red[0] / cols; __syncthreads();
  float vs = 0.0f;
  for (int c = tid; c < cols; c += 256) { float d = vbuf[c] - mean; vs += d * d; }
  red[tid] = vs; __syncthreads();
  for (int o = 128; o > 0; o >>= 1) { if (tid < o) red[tid] += red[tid + o]; __syncthreads(); }
  float rstd = rsqrtf(red[0] / cols + 1e-5f); __syncthreads();
  for (int c = tid; c < cols; c += 256) {
    float y = (vbuf[c] - mean) * rstd * g[c] + b[c];
    if (act == 1) y = gelu_f(y);
    long long idx = (long long)row * cols + c;
    if (out) out[idx] = y;
    if (obf) obf[idx] = f2bf(y);
  }
}

__global__ void softmax_kernel(const float* __restrict__ x, u16* __restrict__ p,
                               int cols, float scale) {
  __shared__ float red[256];
  const long long row = blockIdx.x;
  const int tid = threadIdx.x;
  const float* xr = x + row * cols;
  u16* pr = p + row * cols;
  float mx = -1e30f;
  for (int c = tid; c < cols; c += 256) mx = fmaxf(mx, xr[c] * scale);
  red[tid] = mx; __syncthreads();
  for (int o = 128; o > 0; o >>= 1) { if (tid < o) red[tid] = fmaxf(red[tid], red[tid + o]); __syncthreads(); }
  float MX = red[0]; __syncthreads();
  float s = 0.0f;
  for (int c = tid; c < cols; c += 256) s += expf(xr[c] * scale - MX);
  red[tid] = s; __syncthreads();
  for (int o = 128; o > 0; o >>= 1) { if (tid < o) red[tid] += red[tid + o]; __syncthreads(); }
  float inv = 1.0f / red[0]; __syncthreads();
  for (int c = tid; c < cols; c += 256) pr[c] = f2bf(expf(xr[c] * scale - MX) * inv);
}

// Fused router: logits = x @ w^T + b over NE=8 experts, then softmax.
__global__ void router_kernel(const float* __restrict__ x, const float* __restrict__ w,
                              const float* __restrict__ b, float* __restrict__ rw,
                              int cols) {
  __shared__ float red[256];
  __shared__ float lg[NE];
  const int t = blockIdx.x, tid = threadIdx.x;
  const int wv = tid >> 5, lane = tid & 31;
  const float* xr = x + (long long)t * cols;
  float acc = 0.0f;
  for (int c = lane; c < cols; c += 32) acc += xr[c] * w[wv * cols + c];
  red[tid] = acc; __syncthreads();
  for (int o = 16; o > 0; o >>= 1) { if (lane < o) red[tid] += red[tid + o]; __syncthreads(); }
  if (lane == 0) lg[wv] = red[tid] + b[wv];
  __syncthreads();
  if (tid == 0) {
    float mx = lg[0];
    for (int e = 1; e < NE; ++e) mx = fmaxf(mx, lg[e]);
    float s = 0.0f, ex[NE];
    for (int e = 0; e < NE; ++e) { ex[e] = expf(lg[e] - mx); s += ex[e]; }
    for (int e = 0; e < NE; ++e) rw[(long long)t * NE + e] = ex[e] / s;
  }
}

__global__ void rowdot_sigmoid_kernel(const float* __restrict__ x, const float* __restrict__ w,
                                      const float* __restrict__ b, float* __restrict__ out,
                                      int cols) {
  __shared__ float red[256];
  const int row = blockIdx.x, tid = threadIdx.x;
  const float* xr = x + (long long)row * cols;
  float s = 0.0f;
  for (int c = tid; c < cols; c += 256) s += xr[c] * w[c];
  red[tid] = s; __syncthreads();
  for (int o = 128; o > 0; o >>= 1) { if (tid < o) red[tid] += red[tid + o]; __syncthreads(); }
  if (tid == 0) out[row] = sigmoid_f(red[0] + b[0]);
}

__global__ void concat_kernel(const u16* __restrict__ a, const u16* __restrict__ b,
                              u16* __restrict__ out, long long rows, int cols) {
  long long n = rows * 2 * cols;
  long long i = (long long)blockIdx.x * blockDim.x + threadIdx.x;
  long long st = (long long)gridDim.x * blockDim.x;
  for (; i < n; i += st) {
    long long r = i / (2 * cols); int c = (int)(i % (2 * cols));
    out[i] = (c < cols) ? a[r * cols + c] : b[r * cols + c - cols];
  }
}

// combined = sum_i w_i * r_i with w from cross-verification scores.
__global__ void combine_units_kernel(const float* __restrict__ r0, const float* __restrict__ r1,
                                     const float* __restrict__ r2, const float* __restrict__ s,
                                     float* __restrict__ out, u16* __restrict__ obf,
                                     long long Trows, int cols) {
  long long n = Trows * cols;
  long long i = (long long)blockIdx.x * blockDim.x + threadIdx.x;
  long long st = (long long)gridDim.x * blockDim.x;
  for (; i < n; i += st) {
    long long t = i / cols;
    float s01 = s[t], s02 = s[Trows + t], s12 = s[2 * Trows + t];
    float w0 = (1.0f + s01 + s02) * (1.0f / 3.0f);
    float w1 = (1.0f + s01 + s12) * (1.0f / 3.0f);
    float w2 = (1.0f + s02 + s12) * (1.0f / 3.0f);
    float inv = 1.0f / (w0 + w1 + w2);
    float v = (w0 * r0[i] + w1 * r1[i] + w2 * r2[i]) * inv;
    out[i] = v; obf[i] = f2bf(v);
  }
}

// comp = (a + b) * score[row]
__global__ void addmul_kernel(const float* __restrict__ a, const float* __restrict__ b,
                              const float* __restrict__ sc, float* __restrict__ out,
                              u16* __restrict__ obf, long long n, int cols) {
  long long i = (long long)blockIdx.x * blockDim.x + threadIdx.x;
  long long st = (long long)gridDim.x * blockDim.x;
  for (; i < n; i += st) {
    float v = (a[i] + b[i]) * sc[i / cols];
    out[i] = v; obf[i] = f2bf(v);
  }
}

// X[rows,ld] (head-major cols) -> bf16 [B*NH, S, 64] (tr=0) or [B*NH, 64, S] (tr=1)
__global__ void split_heads_kernel(const float* __restrict__ X, long long ldx,
                                   u16* __restrict__ out, int S, int srcB, int tr) {
  long long n = 32LL * S * DH;
  long long i = (long long)blockIdx.x * blockDim.x + threadIdx.x;
  long long st = (long long)gridDim.x * blockDim.x;
  for (; i < n; i += st) {
    int d, s2; long long bh;
    if (!tr) { d = (int)(i % DH); s2 = (int)((i / DH) % S); bh = i / ((long long)DH * S); }
    else     { s2 = (int)(i % S); d = (int)((i / S) % DH);  bh = i / ((long long)DH * S); }
    int b = (int)(bh >> 4), h = (int)(bh & 15);
    int sb = (srcB == 1) ? 0 : b;
    out[i] = f2bf(X[((long long)sb * S + s2) * ldx + h * DH + d]);
  }
}

// O [B*NH, S, 64] f32 -> bf16 [B*S, H]
__global__ void merge_heads_kernel(const float* __restrict__ O, u16* __restrict__ out, int S) {
  long long n = (long long)2 * S * H;
  long long i = (long long)blockIdx.x * blockDim.x + threadIdx.x;
  long long st = (long long)gridDim.x * blockDim.x;
  for (; i < n; i += st) {
    long long t = i / H; int c = (int)(i % H);
    int b = (int)(t / S), s2 = (int)(t % S), h = c / DH, d = c % DH;
    out[i] = f2bf(O[(((long long)b * NHD + h) * S + s2) * DH + d]);
  }
}

// ---------------------------------------------------------------------------
// Host side
static inline int gs(long long n) {
  long long b = (n + 255) / 256;
  return (int)(b > 8192 ? 8192 : b);
}

static void gemm(hipStream_t st, const u16* A, const u16* Bw, const float* bias,
                 float* C, u16* Cbf, int M_, int N_, int K_,
                 long long sA, long long sB, long long sC, int batch,
                 int act, const float* rs, int rsStride, int accum) {
  dim3 g((N_ + BN - 1) / BN, (M_ + BM - 1) / BM, batch);
  wmma_gemm_kernel<<<g, 256, 0, st>>>(A, Bw, bias, C, Cbf, M_, N_, K_,
                                      sA, sB, sC, act, rs, rsStride, accum);
}

// Input flattening (setup_inputs dict order, recursive insertion order):
//  0 hidden_states
//  units[0]: 1 c1.w 2 c1.b 3 cln.g 4 cln.b 5 c2.w 6 c2.b 7 u1.w 8 u1.b 9 u2.w 10 u2.b 11 s1.w 12 s1.b 13 sln.g 14 sln.b
//  units[1]: 15..24 (c1,cln,c2,u1,u2) 25 s1.w 26 s1.b 27 s2.w 28 s2.b 29 sln.g 30 sln.b
//  units[2]: 31..40 (c1,cln,c2,u1,u2)
//  41 sv1.w 42 sv1.b 43 sv2.w 44 sv2.b 45 cv1.w 46 cv1.b 47 cvln.g 48 cvln.b 49 cv2.w 50 cv2.b
//  51 router.w 52 router.b 53 ew1 54 eb1 55 ew2 56 eb2 57 mem
//  read: 58 in_w 59 in_b 60 out.w 61 out.b   write: 62..65 (unused)  66 gate.w 67 gate.b (unused)
//  layers[l] (l=0,1) base 68+12l: +0 attn.in_w +1 in_b +2 out.w +3 out.b +4 ln1.g +5 ln1.b
//                                 +6 f1.w +7 f1.b +8 f2.w +9 f2.b +10 ln2.g +11 ln2.b
//  92 oi.w 93 oi.b 94 oiln.g 95 oiln.b
extern "C" void kernel_launch(void* const* d_in, const int* in_sizes, int n_in,
                              void* d_out, int out_size, void* d_ws, size_t ws_size,
                              hipStream_t stream) {
  (void)in_sizes; (void)n_in; (void)out_size; (void)ws_size;
  char* base = (char*)d_ws;
  size_t off = 0;
  auto alloc = [&](size_t bytes) -> void* {
    void* p = base + off;
    off = (off + bytes + 255) & ~(size_t)255;
    return p;
  };
  auto F = [&](int i) -> const float* { return (const float*)d_in[i]; };
  auto cvt = [&](const float* src, long long n) -> u16* {
    u16* p = (u16*)alloc((size_t)n * 2);
    cvt_bf16_kernel<<<gs(n), 256, 0, stream>>>(src, p, n);
    return p;
  };

  // ---- bf16 weight (and input) staging ----
  const u16* x_bf = cvt(F(0), (long long)TT * H);
  const u16* u0c1 = cvt(F(1), (long long)H2 * H);
  const u16* u0c2 = cvt(F(5), (long long)H * H2);
  const u16* u0s1 = cvt(F(11), (long long)H * H);
  const u16* u1c1 = cvt(F(15), (long long)H2 * H);
  const u16* u1c2 = cvt(F(19), (long long)H * H2);
  const u16* u1s1 = cvt(F(25), (long long)(H / 2) * H);
  const u16* u1s2 = cvt(F(27), (long long)H * (H / 2));
  const u16* u2c1 = cvt(F(31), (long long)H2 * H);
  const u16* u2c2 = cvt(F(35), (long long)H * H2);
  const u16* sv1w = cvt(F(41), (long long)H * H);
  const u16* cv1w = cvt(F(45), (long long)H * H2);
  const u16* ew1w = cvt(F(53), (long long)NE * H2 * H);
  const u16* ew2w = cvt(F(55), (long long)NE * H * H2);
  const u16* membf = cvt(F(57), (long long)MMEM * H);
  const u16* rinw = cvt(F(58), (long long)H3 * H);
  const u16* routw = cvt(F(60), (long long)H * H);
  const u16* l_ain[2]; const u16* l_aout[2]; const u16* l_f1[2]; const u16* l_f2[2];
  for (int l = 0; l < 2; ++l) {
    int bi = 68 + 12 * l;
    l_ain[l]  = cvt(F(bi + 0), (long long)H3 * H);
    l_aout[l] = cvt(F(bi + 2), (long long)H * H);
    l_f1[l]   = cvt(F(bi + 6), (long long)H4 * H);
    l_f2[l]   = cvt(F(bi + 8), (long long)H * H4);
  }
  const u16* oiw = cvt(F(92), (long long)H * H);

  // ---- activation workspace ----
  float* tA    = (float*)alloc((size_t)TT * H2 * 4);
  u16*   tAbf  = (u16*)alloc((size_t)TT * H2 * 2);
  u16*   cbf   = (u16*)alloc((size_t)TT * H * 2);
  float* t2    = (float*)alloc((size_t)TT * H * 4);
  float* r0    = (float*)alloc((size_t)TT * H * 4);
  u16*   r0b   = (u16*)alloc((size_t)TT * H * 2);
  float* r1    = (float*)alloc((size_t)TT * H * 4);
  u16*   r1b   = (u16*)alloc((size_t)TT * H * 2);
  float* r2    = (float*)alloc((size_t)TT * H * 4);
  u16*   r2b   = (u16*)alloc((size_t)TT * H * 2);
  u16*   halfbf = (u16*)alloc((size_t)TT * (H / 2) * 2);
  u16*   pairbf = (u16*)alloc((size_t)TT * H2 * 2);
  float* vtmp  = (float*)alloc((size_t)TT * H * 4);
  float* vact  = (float*)alloc((size_t)TT * H * 4);
  float* sbuf  = (float*)alloc((size_t)3 * TT * 4);
  float* comb  = (float*)alloc((size_t)TT * H * 4);
  u16*   combbf = (u16*)alloc((size_t)TT * H * 2);
  float* rw    = (float*)alloc((size_t)TT * NE * 4);
  u16*   h1bf  = (u16*)alloc((size_t)TT * H2 * 2);
  float* routed = (float*)alloc((size_t)TT * H * 4);
  u16*   routedbf = (u16*)alloc((size_t)TT * H * 2);
  float* svt   = (float*)alloc((size_t)TT * H * 4);
  float* score = (float*)alloc((size_t)TT * 4);
  float* Qf    = (float*)alloc((size_t)TT * H * 4);
  float* Kf    = (float*)alloc((size_t)MMEM * H * 4);
  float* Vf    = (float*)alloc((size_t)MMEM * H * 4);
  u16*   Qh    = (u16*)alloc((size_t)32 * SS * DH * 2);
  u16*   Kh    = (u16*)alloc((size_t)32 * SS * DH * 2);
  u16*   Vt    = (u16*)alloc((size_t)32 * SS * DH * 2);
  float* scoresb = (float*)alloc((size_t)32 * SS * SS * 4);
  u16*   Pb    = (u16*)alloc((size_t)32 * SS * SS * 2);
  float* Of    = (float*)alloc((size_t)32 * SS * DH * 4);
  u16*   mrg   = (u16*)alloc((size_t)TT * H * 2);
  float* proj  = (float*)alloc((size_t)TT * H * 4);
  float* comp  = (float*)alloc((size_t)TT * H * 4);
  u16*   compbf = (u16*)alloc((size_t)TT * H * 2);
  float* qkv   = (float*)alloc((size_t)TT * H3 * 4);
  u16*   ffnb  = (u16*)alloc((size_t)TT * H4 * 2);
  float* ffn2  = (float*)alloc((size_t)TT * H * 4);

  // ---- Computation units ----
  // unit 0
  gemm(stream, x_bf, u0c1, F(2), tA, nullptr, TT, H2, H, 0, 0, 0, 1, 0, nullptr, 0, 0);
  ln_kernel<<<TT, 256, 0, stream>>>(tA, nullptr, F(3), F(4), nullptr, tAbf, H2, 1);
  gemm(stream, tAbf, u0c2, F(6), nullptr, cbf, TT, H, H2, 0, 0, 0, 1, 0, nullptr, 0, 0);
  gemm(stream, cbf, u0s1, F(12), t2, nullptr, TT, H, H, 0, 0, 0, 1, 0, nullptr, 0, 0);
  ln_kernel<<<TT, 256, 0, stream>>>(t2, nullptr, F(13), F(14), r0, r0b, H, 1);
  // unit 1
  gemm(stream, x_bf, u1c1, F(16), tA, nullptr, TT, H2, H, 0, 0, 0, 1, 0, nullptr, 0, 0);
  ln_kernel<<<TT, 256, 0, stream>>>(tA, nullptr, F(17), F(18), nullptr, tAbf, H2, 1);
  gemm(stream, tAbf, u1c2, F(20), nullptr, cbf, TT, H, H2, 0, 0, 0, 1, 0, nullptr, 0, 0);
  gemm(stream, cbf, u1s1, F(26), nullptr, halfbf, TT, H / 2, H, 0, 0, 0, 1, 1, nullptr, 0, 0);
  gemm(stream, halfbf, u1s2, F(28), t2, nullptr, TT, H, H / 2, 0, 0, 0, 1, 0, nullptr, 0, 0);
  ln_kernel<<<TT, 256, 0, stream>>>(t2, nullptr, F(29), F(30), r1, r1b, H, 0);
  // unit 2
  gemm(stream, x_bf, u2c1, F(32), tA, nullptr, TT, H2, H, 0, 0, 0, 1, 0, nullptr, 0, 0);
  ln_kernel<<<TT, 256, 0, stream>>>(tA, nullptr, F(33), F(34), nullptr, tAbf, H2, 1);
  gemm(stream, tAbf, u2c2, F(36), r2, r2b, TT, H, H2, 0, 0, 0, 1, 0, nullptr, 0, 0);

  // ---- Cross verifier (pairs 01, 02, 12) ----
  const u16* rb[3] = {r0b, r1b, r2b};
  const int pi[3] = {0, 0, 1}, pj[3] = {1, 2, 2};
  for (int p = 0; p < 3; ++p) {
    concat_kernel<<<gs((long long)TT * H2), 256, 0, stream>>>(rb[pi[p]], rb[pj[p]], pairbf, TT, H);
    gemm(stream, pairbf, cv1w, F(46), vtmp, nullptr, TT, H, H2, 0, 0, 0, 1, 0, nullptr, 0, 0);
    ln_kernel<<<TT, 256, 0, stream>>>(vtmp, nullptr, F(47), F(48), vact, nullptr, H, 1);
    rowdot_sigmoid_kernel<<<TT, 256, 0, stream>>>(vact, F(49), F(50), sbuf + (long long)p * TT, H);
  }
  combine_units_kernel<<<gs((long long)TT * H), 256, 0, stream>>>(r0, r1, r2, sbuf, comb, combbf, TT, H);

  // ---- Router + soft MoE ----
  router_kernel<<<TT, 256, 0, stream>>>(comb, F(51), F(52), rw, H);
  zero_kernel<<<gs((long long)TT * H), 256, 0, stream>>>(routed, (long long)TT * H);
  for (int e = 0; e < NE; ++e) {
    gemm(stream, combbf, ew1w + (long long)e * H2 * H, F(54) + (long long)e * H2,
         nullptr, h1bf, TT, H2, H, 0, 0, 0, 1, 1, nullptr, 0, 0);
    gemm(stream, h1bf, ew2w + (long long)e * H * H2, F(56) + (long long)e * H,
         routed, nullptr, TT, H, H2, 0, 0, 0, 1, 0, rw + e, NE, 1);
  }
  cvt_bf16_kernel<<<gs((long long)TT * H), 256, 0, stream>>>(routed, routedbf, (long long)TT * H);

  // ---- Self verifier score ----
  gemm(stream, routedbf, sv1w, F(42), svt, nullptr, TT, H, H, 0, 0, 0, 1, 1, nullptr, 0, 0);
  rowdot_sigmoid_kernel<<<TT, 256, 0, stream>>>(svt, F(43), F(44), score, H);

  // ---- Memory read attention ----
  gemm(stream, routedbf, rinw, F(59), Qf, nullptr, TT, H, H, 0, 0, 0, 1, 0, nullptr, 0, 0);
  gemm(stream, membf, rinw + (long long)H * H, F(59) + H, Kf, nullptr, MMEM, H, H, 0, 0, 0, 1, 0, nullptr, 0, 0);
  gemm(stream, membf, rinw + 2LL * H * H, F(59) + 2 * H, Vf, nullptr, MMEM, H, H, 0, 0, 0, 1, 0, nullptr, 0, 0);
  split_heads_kernel<<<gs(32LL * SS * DH), 256, 0, stream>>>(Qf, H, Qh, SS, 2, 0);
  split_heads_kernel<<<gs(32LL * MMEM * DH), 256, 0, stream>>>(Kf, H, Kh, MMEM, 1, 0);
  split_heads_kernel<<<gs(32LL * MMEM * DH), 256, 0, stream>>>(Vf, H, Vt, MMEM, 1, 1);
  gemm(stream, Qh, Kh, nullptr, scoresb, nullptr, SS, MMEM, DH,
       (long long)SS * DH, (long long)MMEM * DH, (long long)SS * MMEM, 32, 0, nullptr, 0, 0);
  softmax_kernel<<<32 * SS, 256, 0, stream>>>(scoresb, Pb, MMEM, 0.125f);
  gemm(stream, Pb, Vt, nullptr, Of, nullptr, SS, DH, MMEM,
       (long long)SS * MMEM, (long long)DH * MMEM, (long long)SS * DH, 32, 0, nullptr, 0, 0);
  merge_heads_kernel<<<gs((long long)TT * H), 256, 0, stream>>>(Of, mrg, SS);
  gemm(stream, mrg, routw, F(61), proj, nullptr, TT, H, H, 0, 0, 0, 1, 0, nullptr, 0, 0);
  addmul_kernel<<<gs((long long)TT * H), 256, 0, stream>>>(routed, proj, score, comp, compbf,
                                                           (long long)TT * H, H);

  // ---- Transformer composer layers (post-norm, ReLU FFN) ----
  for (int l = 0; l < 2; ++l) {
    int bi = 68 + 12 * l;
    gemm(stream, compbf, l_ain[l], F(bi + 1), qkv, nullptr, TT, H3, H, 0, 0, 0, 1, 0, nullptr, 0, 0);
    split_heads_kernel<<<gs(32LL * SS * DH), 256, 0, stream>>>(qkv, H3, Qh, SS, 2, 0);
    split_heads_kernel<<<gs(32LL * SS * DH), 256, 0, stream>>>(qkv + H, H3, Kh, SS, 2, 0);
    split_heads_kernel<<<gs(32LL * SS * DH), 256, 0, stream>>>(qkv + 2 * H, H3, Vt, SS, 2, 1);
    gemm(stream, Qh, Kh, nullptr, scoresb, nullptr, SS, SS, DH,
         (long long)SS * DH, (long long)SS * DH, (long long)SS * SS, 32, 0, nullptr, 0, 0);
    softmax_kernel<<<32 * SS, 256, 0, stream>>>(scoresb, Pb, SS, 0.125f);
    gemm(stream, Pb, Vt, nullptr, Of, nullptr, SS, DH, SS,
         (long long)SS * SS, (long long)DH * SS, (long long)SS * DH, 32, 0, nullptr, 0, 0);
    merge_heads_kernel<<<gs((long long)TT * H), 256, 0, stream>>>(Of, mrg, SS);
    gemm(stream, mrg, l_aout[l], F(bi + 3), proj, nullptr, TT, H, H, 0, 0, 0, 1, 0, nullptr, 0, 0);
    ln_kernel<<<TT, 256, 0, stream>>>(proj, comp, F(bi + 4), F(bi + 5), comp, compbf, H, 0);
    gemm(stream, compbf, l_f1[l], F(bi + 7), nullptr, ffnb, TT, H4, H, 0, 0, 0, 1, 2, nullptr, 0, 0);
    gemm(stream, ffnb, l_f2[l], F(bi + 9), ffn2, nullptr, TT, H, H4, 0, 0, 0, 1, 0, nullptr, 0, 0);
    ln_kernel<<<TT, 256, 0, stream>>>(ffn2, comp, F(bi + 10), F(bi + 11), comp, compbf, H, 0);
  }

  // ---- Output integration ----
  gemm(stream, compbf, oiw, F(93), proj, nullptr, TT, H, H, 0, 0, 0, 1, 0, nullptr, 0, 0);
  ln_kernel<<<TT, 256, 0, stream>>>(proj, nullptr, F(94), F(95), (float*)d_out, nullptr, H, 0);
}
